// TorchSingleVQ_36215164240445
// MI455X (gfx1250) — matrix operand order repaired and checked by reading, not compiled, not run on hardware
//
#include <hip/hip_runtime.h>
#include <hip/hip_bf16.h>

typedef __attribute__((ext_vector_type(16))) __bf16 v16bf;
typedef __attribute__((ext_vector_type(8)))  float  v8f;

union Frag16 {
    v16bf v;
    unsigned short h[16];
    uint4 q[2];
};

__device__ __forceinline__ unsigned short f2bf_rn(float f) {
    unsigned u = __float_as_uint(f);
    unsigned r = u + 0x7FFFu + ((u >> 16) & 1u);
    return (unsigned short)(r >> 16);
}
__device__ __forceinline__ float bf2f(unsigned short h) {
    return __uint_as_float(((unsigned)h) << 16);
}

// Prepass: codebook fp32 -> bf16 hi/lo split + 0.5*||c||^2 per row.
__global__ __launch_bounds__(128) void vq_prep(const float* __restrict__ cb,
                                               unsigned short* __restrict__ cbh,
                                               unsigned short* __restrict__ cbl,
                                               float* __restrict__ csqh) {
    const int row = blockIdx.x;
    const int d   = threadIdx.x;
    float f = cb[(size_t)row * 128 + d];
    unsigned short hh = f2bf_rn(f);
    float fl = f - bf2f(hh);
    cbh[(size_t)row * 128 + d] = hh;
    cbl[(size_t)row * 128 + d] = f2bf_rn(fl);

    __shared__ float red[128];
    red[d] = f * f;
    __syncthreads();
    for (int s = 64; s > 0; s >>= 1) {
        if (d < s) red[d] += red[d + s];
        __syncthreads();
    }
    if (d == 0) csqh[row] = 0.5f * red[0];
}

// Main: each wave owns 16 rows; loops over K in 16-code chunks with
// split-bf16 WMMA. Three independent accumulator chains (xh*ch, xh*cl, xl*ch)
// expose 3-way ILP on the matrix pipe instead of one serial 12-deep chain.
__global__ __launch_bounds__(256) void vq_main(const float* __restrict__ x,
                                               const float* __restrict__ cb,
                                               const unsigned short* __restrict__ cbh,
                                               const unsigned short* __restrict__ cbl,
                                               const float* __restrict__ csqh,
                                               float* __restrict__ quant,
                                               int* __restrict__ codes,
                                               int Btot, int Ktot) {
    const int lane  = threadIdx.x & 31;
    const int wave  = threadIdx.x >> 5;
    const int rowBase = blockIdx.x * 128 + wave * 16;
    const int r     = lane & 15;          // row / code within 16-slice
    const int khalfA = (lane >> 4) * 16;  // A layout: K base 0 or 16
    const int ksetB  = (lane >> 4) * 16;  // B layout: K base 0 or 16

    // ---- Load A fragments (16 rows x 128 dims) as bf16 hi/lo, resident ----
    int row = rowBase + r;
    if (row >= Btot) row = Btot - 1;      // clamp: EXEC must stay all-ones for WMMA
    const float* xr = x + (size_t)row * 128;

    Frag16 ah[4], al[4];
#pragma unroll
    for (int kf = 0; kf < 4; ++kf) {
        const int kb = kf * 32 + khalfA;
#pragma unroll
        for (int g = 0; g < 2; ++g) {     // elems 0..7 -> K=kb..kb+7 ; 8..15 -> K=kb+8..kb+15
#pragma unroll
            for (int e = 0; e < 8; ++e) {
                float f = xr[kb + g * 8 + e];
                unsigned short hh = f2bf_rn(f);
                float fl = f - bf2f(hh);
                ah[kf].h[g * 8 + e] = hh;
                al[kf].h[g * 8 + e] = f2bf_rn(fl);
            }
        }
    }

    float bestv[8];
    int   besti[8];
#pragma unroll
    for (int i = 0; i < 8; ++i) { bestv[i] = 3.4e38f; besti[i] = 0; }

    const int nchunks = Ktot >> 4;
    for (int chunk = 0; chunk < nchunks; ++chunk) {
        const int nb = chunk * 16;
        const float hterm = csqh[nb + r];        // 0.5*||c||^2 for this lane's code
        v8f accHH = {0.f, 0.f, 0.f, 0.f, 0.f, 0.f, 0.f, 0.f};
        v8f accHL = accHH;
        v8f accLH = accHH;
#pragma unroll
        for (int kf = 0; kf < 4; ++kf) {
            // B fragment: 32(K) x 16(N); lane holds 16 contiguous bf16 of code row (nb+r)
            const size_t boff = (size_t)(nb + r) * 128 + kf * 32 + ksetB;
            Frag16 bh, bl;
            bh.q[0] = *(const uint4*)(cbh + boff);
            bh.q[1] = *(const uint4*)(cbh + boff + 8);
            bl.q[0] = *(const uint4*)(cbl + boff);
            bl.q[1] = *(const uint4*)(cbl + boff + 8);
            // Three independent accumulation chains -> 3-way matrix-pipe ILP.
            accHH = __builtin_amdgcn_wmma_f32_16x16x32_bf16(false, ah[kf].v, false, bh.v,
                                                            (short)0, accHH, false, false);
            accHL = __builtin_amdgcn_wmma_f32_16x16x32_bf16(false, ah[kf].v, false, bl.v,
                                                            (short)0, accHL, false, false);
            accLH = __builtin_amdgcn_wmma_f32_16x16x32_bf16(false, al[kf].v, false, bh.v,
                                                            (short)0, accLH, false, false);
        }
        const int n = nb + r;
#pragma unroll
        for (int i = 0; i < 8; ++i) {
            float dot = accHH[i] + (accHL[i] + accLH[i]);
            float s = hterm - dot;               // argmin(||c||^2/2 - x.c) == argmin dist
            bool lt = s < bestv[i];              // strict '<' keeps earliest (smallest) index
            bestv[i] = lt ? s : bestv[i];
            besti[i] = lt ? n : besti[i];
        }
    }

    // ---- Reduce across the 16 codes held in each lane-half (butterfly) ----
#pragma unroll
    for (int i = 0; i < 8; ++i) {
        float v = bestv[i];
        int   ix = besti[i];
#pragma unroll
        for (int off = 1; off < 16; off <<= 1) {
            float ov = __shfl_xor(v, off, 32);
            int   oi = __shfl_xor(ix, off, 32);
            if (ov < v || (ov == v && oi < ix)) { v = ov; ix = oi; }
        }
        bestv[i] = v; besti[i] = ix;
    }

    // ---- Emit codes + gather codebook rows (fp32 source -> exact output) ----
#pragma unroll
    for (int i = 0; i < 8; ++i) {
        const int rowA = rowBase + i;        // C/D layout: VGPR i, lanes 0-15  -> row i
        const int rowB = rowBase + i + 8;    //             VGPR i, lanes 16-31 -> row i+8
        const int idxA = __shfl(besti[i], 0, 32);
        const int idxB = __shfl(besti[i], 16, 32);
        if (rowA < Btot) {
            if (lane == 0) codes[rowA] = idxA;
            const float4* src = (const float4*)(cb + (size_t)idxA * 128);
            float4*       dst = (float4*)(quant + (size_t)rowA * 128);
            dst[lane] = src[lane];           // 32 lanes x float4 = 128 floats
        }
        if (rowB < Btot) {
            if (lane == 0) codes[rowB] = idxB;
            const float4* src = (const float4*)(cb + (size_t)idxB * 128);
            float4*       dst = (float4*)(quant + (size_t)rowB * 128);
            dst[lane] = src[lane];
        }
    }
}

extern "C" void kernel_launch(void* const* d_in, const int* in_sizes, int n_in,
                              void* d_out, int out_size, void* d_ws, size_t ws_size,
                              hipStream_t stream) {
    const float* x  = (const float*)d_in[0];
    const float* cb = (const float*)d_in[1];
    const int D = 128;
    const int B = in_sizes[0] / D;
    const int K = in_sizes[1] / D;

    unsigned short* cbh  = (unsigned short*)d_ws;
    unsigned short* cbl  = cbh + (size_t)K * D;
    float*          csqh = (float*)(cbl + (size_t)K * D);

    float* quant = (float*)d_out;
    int*   codes = (int*)(quant + (size_t)B * D);

    vq_prep<<<K, 128, 0, stream>>>(cb, cbh, cbl, csqh);

    const int rowsPerBlock = 128;            // 8 waves x 16 rows
    const int grid = (B + rowsPerBlock - 1) / rowsPerBlock;
    vq_main<<<grid, 256, 0, stream>>>(x, cb, cbh, cbl, csqh, quant, codes, B, K);
}
